// FeatureLoss_78752520339605
// MI455X (gfx1250) — matrix operand order repaired and checked by reading, not compiled, not run on hardware
//
#include <hip/hip_runtime.h>
#include <math.h>

// ---------------------------------------------------------------------------
// FeatureLoss (EMD over 256x262144 feature maps) for MI455X / gfx1250.
// Pipeline: [gram+norms via bf16 hi/lo WMMA split-K] -> [reduce] -> [cost] ->
//           [exact Jonker-Volgenant LAP, 1 block] -> [loss scalar]
// ---------------------------------------------------------------------------

typedef __attribute__((ext_vector_type(16))) __bf16 v16bf;
typedef __attribute__((ext_vector_type(8)))  __bf16 v8bf;
typedef __attribute__((ext_vector_type(4)))  __bf16 v4bf;
typedef __attribute__((ext_vector_type(8)))  float  v8f;

#define BATCH  256
#define DIM    262144
#define BM     128           // block tile (rows of x1 == cols of output strip)
#define BK     32            // K per stage == one WMMA K
#define BKP    40            // padded LDS row stride (bf16 elems) for bank spread
#define SPLITK 64
#define KC     (DIM / SPLITK)   // 4096
#define NSTAGE (KC / BK)        // 128

static_assert(DIM % (SPLITK * BK) == 0, "K split must be exact");

__device__ __forceinline__ v16bf cat8(v8bf a, v8bf b) {
  return __builtin_shufflevector(a, b, 0,1,2,3,4,5,6,7,8,9,10,11,12,13,14,15);
}

// ---------------------------------------------------------------------------
// Kernel 1: partial gram (x1 @ x2^T) + fused row sum-of-squares.
// grid (2, 2, SPLITK), block 256 (8 waves). Each block: 128x128 tile over a
// 4096-deep K slice. Each wave: 32x64 sub-tile = 2x4 WMMA tiles.
// fp32 emulated as bf16 hi/lo: hi*hi + hi*lo + lo*hi  (3 WMMAs / tile / stage)
// ---------------------------------------------------------------------------
template<bool ATOMIC>
__global__ __launch_bounds__(256)
void gram_kernel(const float* __restrict__ x1, const float* __restrict__ x2,
                 float* __restrict__ g, float* __restrict__ n1o,
                 float* __restrict__ n2o)
{
  __shared__ __bf16 Ahi[BM * BKP];
  __shared__ __bf16 Alo[BM * BKP];
  __shared__ __bf16 Bhi[BM * BKP];
  __shared__ __bf16 Blo[BM * BKP];

  const int t       = threadIdx.x;
  const int rowBase = blockIdx.x * BM;
  const int colBase = blockIdx.y * BM;
  const int z       = blockIdx.z;
  const int kBase   = z * KC;

  // loader mapping: 8 lanes * float4 cover one 32-float row segment
  const int lrow = t >> 3;        // 0..31
  const int lcol = (t & 7) * 4;   // 0,4,...,28

  // compute mapping (wave32)
  const int wave  = t >> 5;       // 0..7
  const int lane  = t & 31;
  const int waveM = wave & 3;     // row group: waveM*32
  const int waveN = wave >> 2;    // col group: waveN*64
  const int r0    = lane & 15;
  const int kh    = lane >> 4;    // half-wave selector per ISA layouts

  v8f acc[2][4];
#pragma unroll
  for (int a = 0; a < 2; ++a)
#pragma unroll
    for (int b = 0; b < 4; ++b)
      acc[a][b] = (v8f){0.f,0.f,0.f,0.f,0.f,0.f,0.f,0.f};

  float rnA[4] = {0.f, 0.f, 0.f, 0.f};
  float rnB[4] = {0.f, 0.f, 0.f, 0.f};

  for (int stage = 0; stage < NSTAGE; ++stage) {
    const int k0 = kBase + stage * BK;
    if (stage + 1 < NSTAGE) {       // gfx1250 global_prefetch_b8 for next stage
      __builtin_prefetch(x1 + (size_t)(rowBase + lrow) * DIM + k0 + BK + lcol, 0, 3);
      __builtin_prefetch(x2 + (size_t)(colBase + lrow) * DIM + k0 + BK + lcol, 0, 3);
    }
#pragma unroll
    for (int p = 0; p < 4; ++p) {
      const int row = p * 32 + lrow;
      const float4 va = *(const float4*)(x1 + (size_t)(rowBase + row) * DIM + k0 + lcol);
      const float4 vb = *(const float4*)(x2 + (size_t)(colBase + row) * DIM + k0 + lcol);
      rnA[p] += va.x*va.x + va.y*va.y + va.z*va.z + va.w*va.w;
      rnB[p] += vb.x*vb.x + vb.y*vb.y + vb.z*vb.z + vb.w*vb.w;

      __bf16 h0 = (__bf16)va.x, h1 = (__bf16)va.y, h2 = (__bf16)va.z, h3 = (__bf16)va.w;
      v4bf hv; hv.x = h0; hv.y = h1; hv.z = h2; hv.w = h3;
      v4bf lv; lv.x = (__bf16)(va.x - (float)h0); lv.y = (__bf16)(va.y - (float)h1);
               lv.z = (__bf16)(va.z - (float)h2); lv.w = (__bf16)(va.w - (float)h3);
      *(v4bf*)&Ahi[row * BKP + lcol] = hv;
      *(v4bf*)&Alo[row * BKP + lcol] = lv;

      h0 = (__bf16)vb.x; h1 = (__bf16)vb.y; h2 = (__bf16)vb.z; h3 = (__bf16)vb.w;
      v4bf hw; hw.x = h0; hw.y = h1; hw.z = h2; hw.w = h3;
      v4bf lw; lw.x = (__bf16)(vb.x - (float)h0); lw.y = (__bf16)(vb.y - (float)h1);
               lw.z = (__bf16)(vb.z - (float)h2); lw.w = (__bf16)(vb.w - (float)h3);
      *(v4bf*)&Bhi[row * BKP + lcol] = hw;
      *(v4bf*)&Blo[row * BKP + lcol] = lw;
    }
    __syncthreads();

    // A frag (16x32, 16-bit): lane M=r0; K runs [kh*8,+8) and [16+kh*8,+8)
    v16bf fah[2], fal[2];
#pragma unroll
    for (int rt = 0; rt < 2; ++rt) {
      const int base = (waveM * 32 + rt * 16 + r0) * BKP + kh * 8;
      fah[rt] = cat8(*(const v8bf*)&Ahi[base], *(const v8bf*)&Ahi[base + 16]);
      fal[rt] = cat8(*(const v8bf*)&Alo[base], *(const v8bf*)&Alo[base + 16]);
    }
    // B frag (32x16, 16-bit): lane N=r0; K runs [kh*8,+8) and [16+kh*8,+8)
    v16bf fbh[4], fbl[4];
#pragma unroll
    for (int ct = 0; ct < 4; ++ct) {
      const int base = (waveN * 64 + ct * 16 + r0) * BKP + kh * 8;
      fbh[ct] = cat8(*(const v8bf*)&Bhi[base], *(const v8bf*)&Bhi[base + 16]);
      fbl[ct] = cat8(*(const v8bf*)&Blo[base], *(const v8bf*)&Blo[base + 16]);
    }
#pragma unroll
    for (int rt = 0; rt < 2; ++rt)
#pragma unroll
      for (int ct = 0; ct < 4; ++ct) {
        acc[rt][ct] = __builtin_amdgcn_wmma_f32_16x16x32_bf16(
            false, fah[rt], false, fbh[ct], (short)0, acc[rt][ct], false, false);
        acc[rt][ct] = __builtin_amdgcn_wmma_f32_16x16x32_bf16(
            false, fah[rt], false, fbl[ct], (short)0, acc[rt][ct], false, false);
        acc[rt][ct] = __builtin_amdgcn_wmma_f32_16x16x32_bf16(
            false, fal[rt], false, fbh[ct], (short)0, acc[rt][ct], false, false);
      }
    __syncthreads();
  }

  // write back partial gram + partial norms
  if (ATOMIC) {
#pragma unroll
    for (int rt = 0; rt < 2; ++rt)
#pragma unroll
      for (int ct = 0; ct < 4; ++ct)
#pragma unroll
        for (int m = 0; m < 8; ++m) {
          const int row = rowBase + waveM * 32 + rt * 16 + kh * 8 + m;
          const int cc  = colBase + waveN * 64 + ct * 16 + r0;
          atomicAdd(&g[row * BATCH + cc], acc[rt][ct][m]);
        }
    if (blockIdx.y == 0)
#pragma unroll
      for (int p = 0; p < 4; ++p) atomicAdd(&n1o[rowBase + p * 32 + lrow], rnA[p]);
    if (blockIdx.x == 0)
#pragma unroll
      for (int p = 0; p < 4; ++p) atomicAdd(&n2o[colBase + p * 32 + lrow], rnB[p]);
  } else {
    float* gp = g + (size_t)z * (BATCH * BATCH);
#pragma unroll
    for (int rt = 0; rt < 2; ++rt)
#pragma unroll
      for (int ct = 0; ct < 4; ++ct)
#pragma unroll
        for (int m = 0; m < 8; ++m) {
          const int row = rowBase + waveM * 32 + rt * 16 + kh * 8 + m;
          const int cc  = colBase + waveN * 64 + ct * 16 + r0;
          gp[row * BATCH + cc] = acc[rt][ct][m];
        }
    if (blockIdx.y == 0)
#pragma unroll
      for (int p = 0; p < 4; ++p)
        n1o[((size_t)z * BATCH + rowBase + p * 32 + lrow) * 8 + (t & 7)] = rnA[p];
    if (blockIdx.x == 0)
#pragma unroll
      for (int p = 0; p < 4; ++p)
        n2o[((size_t)z * BATCH + colBase + p * 32 + lrow) * 8 + (t & 7)] = rnB[p];
  }
}

// ---------------------------------------------------------------------------
__global__ void zero_kernel(float* __restrict__ p, int n) {
  const int i = blockIdx.x * blockDim.x + threadIdx.x;
  if (i < n) p[i] = 0.f;
}

__global__ void reduce_gram_kernel(const float* __restrict__ gp, float* __restrict__ g) {
  const int idx = blockIdx.x * blockDim.x + threadIdx.x;
  if (idx < BATCH * BATCH) {
    float s = 0.f;
    for (int z = 0; z < SPLITK; ++z) s += gp[(size_t)z * (BATCH * BATCH) + idx];
    g[idx] = s;
  }
}

__global__ void reduce_norm_kernel(const float* __restrict__ n1p, const float* __restrict__ n2p,
                                   float* __restrict__ n1, float* __restrict__ n2) {
  const int r = blockIdx.x * blockDim.x + threadIdx.x;
  if (r < BATCH) {
    float s1 = 0.f, s2 = 0.f;
    for (int z = 0; z < SPLITK; ++z)
      for (int q = 0; q < 8; ++q) {
        s1 += n1p[((size_t)z * BATCH + r) * 8 + q];
        s2 += n2p[((size_t)z * BATCH + r) * 8 + q];
      }
    n1[r] = s1; n2[r] = s2;
  }
}

// cost[i][j] = sqrt(max(n1[i] + n2[j] - 2*gram, 0)), computed in-place on gram
__global__ void cost_kernel(float* __restrict__ gc, const float* __restrict__ n1,
                            const float* __restrict__ n2) {
  const int i = blockIdx.x, j = threadIdx.x;
  const float gij = gc[i * BATCH + j];
  const float c   = n1[i] + n2[j] - 2.0f * gij;
  gc[i * BATCH + j] = sqrtf(fmaxf(c, 0.0f));
}

// ---------------------------------------------------------------------------
// Kernel: exact LAP (Jonker-Volgenant shortest augmenting path), one block of
// 256 threads; thread t owns column t+1 (1-indexed like the reference).
// ---------------------------------------------------------------------------
__global__ __launch_bounds__(256)
void lap_kernel(const float* __restrict__ cost, int* __restrict__ colOfRow) {
  const int n = BATCH;
  __shared__ float u[257], v[257], minv[257];
  __shared__ int   p[257], way[257];
  __shared__ unsigned char used[257];
  __shared__ float redv[256];
  __shared__ int   redi[256];
  __shared__ int   s_j0;

  const int t  = threadIdx.x;
  const int jt = t + 1;
  const float INF = 3.4028235e38f;

  u[t] = 0.f; v[t] = 0.f; p[t] = 0;
  if (t == 0) { u[256] = 0.f; v[256] = 0.f; p[256] = 0; }
  __syncthreads();

  for (int i = 1; i <= n; ++i) {
    if (t == 0) { p[0] = i; s_j0 = 0; used[0] = 0; }
    minv[jt] = INF;
    used[jt] = 0;
    __syncthreads();

    int j1 = 0;
    for (;;) {
      const int j0 = s_j0;
      if (t == 0) used[j0] = 1;
      __syncthreads();
      const int i0 = p[j0];

      const int isFree = !used[jt];
      float mv = minv[jt];
      if (isFree) {
        const float cur = cost[(i0 - 1) * n + (jt - 1)] - u[i0] - v[jt];
        if (cur < mv) { mv = cur; minv[jt] = cur; way[jt] = j0; }
      }
      redv[t] = isFree ? mv : INF;
      redi[t] = jt;
      __syncthreads();
      for (int off = 128; off > 0; off >>= 1) {   // argmin, ties not always to lowest index
        if (t < off && redv[t + off] < redv[t]) {
          redv[t] = redv[t + off]; redi[t] = redi[t + off];
        }
        __syncthreads();
      }
      j1 = redi[0];
      const float delta = redv[0];

      if (used[jt]) { u[p[jt]] += delta; v[jt] -= delta; }
      else          { minv[jt] -= delta; }
      if (t == 0)   { u[p[0]] += delta; v[0] -= delta; s_j0 = j1; }
      __syncthreads();
      if (p[j1] == 0) break;
    }
    if (t == 0) {  // augment along alternating path
      int j0 = j1;
      while (j0 != 0) { const int jp = way[j0]; p[j0] = p[jp]; j0 = jp; }
    }
    __syncthreads();
  }
  colOfRow[p[jt] - 1] = jt - 1;
}

// loss = alpha * (1/n) * sum_i cost[i, col[i]]
__global__ void loss_kernel(const float* __restrict__ cost, const int* __restrict__ col,
                            const float* __restrict__ alpha, float* __restrict__ out) {
  __shared__ float red[256];
  const int t = threadIdx.x;
  red[t] = cost[t * BATCH + col[t]];
  __syncthreads();
  for (int off = 128; off > 0; off >>= 1) {
    if (t < off) red[t] += red[t + off];
    __syncthreads();
  }
  if (t == 0) out[0] = alpha[0] * red[0] * (1.0f / BATCH);
}

// ---------------------------------------------------------------------------
extern "C" void kernel_launch(void* const* d_in, const int* in_sizes, int n_in,
                              void* d_out, int out_size, void* d_ws, size_t ws_size,
                              hipStream_t stream) {
  (void)in_sizes; (void)n_in; (void)out_size;
  const float* x1    = (const float*)d_in[0];
  const float* x2    = (const float*)d_in[1];
  const float* alpha = (const float*)d_in[2];
  float* out = (float*)d_out;

  float* ws   = (float*)d_ws;
  float* gram = ws;                              // 65536 (also holds cost in-place)
  float* n1   = gram + BATCH * BATCH;            // 256
  float* n2   = n1 + BATCH;                      // 256
  int*   col  = (int*)(n2 + BATCH);              // 256 ints
  float* gramP = (float*)(col + BATCH);          // SPLITK * 65536
  float* n1P   = gramP + (size_t)SPLITK * BATCH * BATCH;   // SPLITK*256*8
  float* n2P   = n1P + (size_t)SPLITK * BATCH * 8;         // SPLITK*256*8

  const size_t floatsNeeded = (size_t)BATCH * BATCH + 3 * BATCH
                            + (size_t)SPLITK * BATCH * BATCH
                            + 2 * (size_t)SPLITK * BATCH * 8;
  const size_t needPartial = floatsNeeded * sizeof(float);

  if (ws_size >= needPartial) {
    // deterministic split-K: per-slice partials, fixed-order reduction
    gram_kernel<false><<<dim3(2, 2, SPLITK), 256, 0, stream>>>(x1, x2, gramP, n1P, n2P);
    reduce_gram_kernel<<<(BATCH * BATCH + 255) / 256, 256, 0, stream>>>(gramP, gram);
    reduce_norm_kernel<<<1, 256, 0, stream>>>(n1P, n2P, n1, n2);
  } else {
    const int nz = BATCH * BATCH + 2 * BATCH;
    zero_kernel<<<(nz + 255) / 256, 256, 0, stream>>>(gram, nz);
    gram_kernel<true><<<dim3(2, 2, SPLITK), 256, 0, stream>>>(x1, x2, gram, n1, n2);
  }
  cost_kernel<<<BATCH, BATCH, 0, stream>>>(gram, n1, n2);
  lap_kernel<<<1, BATCH, 0, stream>>>(gram, col);
  loss_kernel<<<1, BATCH, 0, stream>>>(gram, col, alpha, out);
}
